// CVRPModel_45655502356508
// MI455X (gfx1250) — compile-verified, matrix operand order checked by reading, not run on hardware
//
#include <hip/hip_runtime.h>
#include <hip/hip_bf16.h>
#include <math.h>

#define B_    512
#define N_    100
#define E_    128
#define H_    8
#define DH_   16
#define FF_   2048
#define BN_   (B_*N_)
#define NEG_  (-1000000000.0f)

typedef __attribute__((ext_vector_type(16))) _Float16 v16h;
typedef __attribute__((ext_vector_type(8)))  _Float16 v8h;
typedef __attribute__((ext_vector_type(8)))  float    v8f;
typedef __attribute__((ext_vector_type(4)))  unsigned int u32x4;
typedef __attribute__((ext_vector_type(8)))  int i32x8;
typedef __attribute__((ext_vector_type(4)))  int i32x4;

#if defined(__gfx1250__) && __has_builtin(__builtin_amdgcn_tensor_load_to_lds)
#define USE_TDM 1
#else
#define USE_TDM 0
#endif

// ---------------------------------------------------------------------------
// WMMA helpers: CDNA5 16x16x32 f16 -> f32.
// A fragment (16xK slice, row-major, lane = M index): lane l holds row (l&15);
//   element j of v16h is K = k0 + (j>=8 ? 16 : 0) + (l>>4)*8 + (j&7)
//   -> two contiguous 16B runs per lane. B fragment mirrors this with
//   lane = N index over W[N,K] rows (W row n == column n of B = KxN).
// ---------------------------------------------------------------------------
__device__ __forceinline__ v16h ld_frag(const _Float16* base, int stride, int k0,
                                        int ml, int hi) {
  const _Float16* p = base + (size_t)ml * stride + k0 + hi * 8;
  v8h lo = *(const v8h*)p;
  v8h hh = *(const v8h*)(p + 16);
  v16h r;
#pragma unroll
  for (int j = 0; j < 8; ++j) { r[j] = lo[j]; r[j + 8] = hh[j]; }
  return r;
}

__device__ __forceinline__ v8f wmma16(v16h a, v16h b, v8f c) {
  return __builtin_amdgcn_wmma_f32_16x16x32_f16(false, a, false, b, (short)0, c,
                                                false, false);
}

#if USE_TDM
// ---------------------------------------------------------------------------
// Tensor Data Mover: 2D tile (fp16 elements) global -> LDS.
// D# per CDNA5 ISA sec.8:
//  group0: [1:0]=count=1, [63:32]=lds_addr, [120:64]=global_addr, [127:126]=2
//  group1: [17:16]=data_size(1->2B), [79:48]=tensor_dim0, [111:80]=tensor_dim1,
//          [127:112]=tile_dim0, [143:128]=tile_dim1, [207:160]=dim0_stride
// Issue from one wave only (TDM ignores EXEC); fence with s_wait_tensorcnt.
// ---------------------------------------------------------------------------
__device__ __forceinline__ void tdm_load_2d_f16(unsigned lds_addr, const void* gptr,
                                                unsigned tile_d0, unsigned tile_d1,
                                                unsigned tens_d0, unsigned tens_d1,
                                                unsigned stride0) {
  unsigned long long ga = (unsigned long long)(uintptr_t)gptr;
  u32x4 g0;
  g0[0] = 1u;                                                // count=1 (user)
  g0[1] = lds_addr;                                          // LDS byte addr
  g0[2] = (unsigned)(ga & 0xffffffffu);                      // global_addr lo
  g0[3] = (unsigned)((ga >> 32) & 0x01ffffffu) | (2u << 30); // addr hi | type=2
  i32x8 g1;
  g1[0] = (int)(1u << 16);                                   // data_size = 2B
  g1[1] = (int)((tens_d0 & 0xffffu) << 16);                  // tensor_dim0 lo16
  g1[2] = (int)(((tens_d0 >> 16) & 0xffffu) | ((tens_d1 & 0xffffu) << 16));
  g1[3] = (int)(((tens_d1 >> 16) & 0xffffu) | ((tile_d0 & 0xffffu) << 16));
  g1[4] = (int)(tile_d1 & 0xffffu);                          // tile_dim1, d2=0
  g1[5] = (int)stride0;                                      // dim0 stride lo
  g1[6] = 0;
  g1[7] = 0;
  i32x4 z4 = {};
#if __clang_major__ >= 23
  i32x8 z8 = {};
  __builtin_amdgcn_tensor_load_to_lds(g0, g1, z4, z4, z8, 0);
#else
  __builtin_amdgcn_tensor_load_to_lds(g0, g1, z4, z4, 0);
#endif
}
#endif

// ---------------------------------------------------------------------------
// fp32 -> fp16 weight conversion
// ---------------------------------------------------------------------------
__global__ void cvt_f16_kernel(const float* __restrict__ src,
                               _Float16* __restrict__ dst, int n) {
  int i = blockIdx.x * 256 + threadIdx.x;
  if (i < n) dst[i] = (_Float16)src[i];
}

// ---------------------------------------------------------------------------
// Embedding: h = [cx, cy, demand/cap] @ embed_W.T + embed_b ; also emits dr.
// ---------------------------------------------------------------------------
__global__ void embed_kernel(const float* __restrict__ coords,
                             const float* __restrict__ demands,
                             const float* __restrict__ capacity,
                             const float* __restrict__ eW,
                             const float* __restrict__ eb,
                             float* __restrict__ h32,
                             _Float16* __restrict__ h16,
                             float* __restrict__ dr_out) {
  int idx = blockIdx.x * 256 + threadIdx.x;
  if (idx >= BN_ * E_) return;
  int row = idx >> 7, e = idx & 127;
  int b = row / N_;
  float dr = demands[row] / capacity[b];
  float cx = coords[(size_t)row * 2 + 0];
  float cy = coords[(size_t)row * 2 + 1];
  float v = eW[e * 3 + 0] * cx + eW[e * 3 + 1] * cy + eW[e * 3 + 2] * dr + eb[e];
  h32[idx] = v;
  h16[idx] = (_Float16)v;
  if (e == 0) dr_out[row] = dr;
}

// ---------------------------------------------------------------------------
// Generic WMMA GEMM: C[M,N] = act(A16[M,K] @ W16[N,K]^T + bias)
// Block = 256 threads (8 waves). Wave owns 16 rows x 64 cols (4 n-tiles).
// All 5 fragments of a K-step are issued before the 4 WMMAs so the loads
// batch into one clause with a single s_wait_loadcnt.
// ---------------------------------------------------------------------------
__global__ void gemm_kernel(const _Float16* __restrict__ A,
                            const _Float16* __restrict__ W,
                            const float* __restrict__ bias,
                            float* __restrict__ out32,
                            _Float16* __restrict__ out16,
                            int N, int K, int relu) {
  int tid = threadIdx.x, wid = tid >> 5, lane = tid & 31;
  int ml = lane & 15, hi = lane >> 4;
  size_t m0 = (size_t)blockIdx.x * 128 + wid * 16;
  int n0 = blockIdx.y * 64;
  const _Float16* Abase = A + m0 * K;
  const _Float16* W0 = W + (size_t)(n0 +  0) * K;
  const _Float16* W1 = W + (size_t)(n0 + 16) * K;
  const _Float16* W2 = W + (size_t)(n0 + 32) * K;
  const _Float16* W3 = W + (size_t)(n0 + 48) * K;
  v8f acc[4] = {};
  for (int k0 = 0; k0 < K; k0 += 32) {
    __builtin_prefetch(Abase + (size_t)ml * K + k0 + 64, 0, 1);
    v16h a  = ld_frag(Abase, K, k0, ml, hi);
    v16h b0 = ld_frag(W0, K, k0, ml, hi);
    v16h b1 = ld_frag(W1, K, k0, ml, hi);
    v16h b2 = ld_frag(W2, K, k0, ml, hi);
    v16h b3 = ld_frag(W3, K, k0, ml, hi);
    acc[0] = wmma16(a, b0, acc[0]);
    acc[1] = wmma16(a, b1, acc[1]);
    acc[2] = wmma16(a, b2, acc[2]);
    acc[3] = wmma16(a, b3, acc[3]);
  }
#pragma unroll
  for (int t = 0; t < 4; ++t)
#pragma unroll
    for (int r = 0; r < 8; ++r) {
      size_t m = m0 + r + hi * 8;
      int n = n0 + t * 16 + ml;
      float v = acc[t][r] + (bias ? bias[n] : 0.0f);
      if (relu) v = fmaxf(v, 0.0f);
      if (out32) out32[m * (size_t)N + n] = v;
      if (out16) out16[m * (size_t)N + n] = (_Float16)v;
    }
}

// ---------------------------------------------------------------------------
// Attention: one block per (b, head). dh=16, N=100.
// q/k/v tiles (100x16 fp16, row stride 384) pulled into LDS by the TDM.
// ---------------------------------------------------------------------------
__global__ void attn_kernel(const _Float16* __restrict__ qkv,
                            _Float16* __restrict__ attn_out) {
  int bh = blockIdx.x;
  int b = bh >> 3, h = bh & 7;
  int tid = threadIdx.x, wid = tid >> 5, lane = tid & 31;
  __shared__ _Float16 sQ[N_][DH_], sK[N_][DH_], sV[N_][DH_];
  __shared__ float sS[N_][N_];
  const _Float16* base = qkv + (size_t)b * N_ * (3 * E_);
#if USE_TDM
  if (tid < 32) {  // wave 0 issues three 2D tensor DMAs
    tdm_load_2d_f16((unsigned)(uintptr_t)&sQ[0][0], base + h * 16,
                    DH_, N_, 3 * E_, N_, 3 * E_);
    tdm_load_2d_f16((unsigned)(uintptr_t)&sK[0][0], base + E_ + h * 16,
                    DH_, N_, 3 * E_, N_, 3 * E_);
    tdm_load_2d_f16((unsigned)(uintptr_t)&sV[0][0], base + 2 * E_ + h * 16,
                    DH_, N_, 3 * E_, N_, 3 * E_);
    __builtin_amdgcn_s_wait_tensorcnt(0);
  }
  __syncthreads();
#else
  for (int i = tid; i < N_ * DH_; i += 256) {
    int n = i >> 4, d = i & 15;
    const _Float16* row = base + (size_t)n * (3 * E_);
    sQ[n][d] = row[h * 16 + d];
    sK[n][d] = row[E_ + h * 16 + d];
    sV[n][d] = row[2 * E_ + h * 16 + d];
  }
  __syncthreads();
#endif
  for (int i = tid; i < N_ * N_; i += 256) {
    int r = i / N_, c = i % N_;
    float s = 0.f;
#pragma unroll
    for (int d = 0; d < DH_; ++d) s += (float)sQ[r][d] * (float)sK[c][d];
    sS[r][c] = s * 0.25f;  // 1/sqrt(16)
  }
  __syncthreads();
  for (int r = wid; r < N_; r += 8) {
    float m = -3.4e38f;
    for (int c = lane; c < N_; c += 32) m = fmaxf(m, sS[r][c]);
    for (int off = 16; off > 0; off >>= 1) m = fmaxf(m, __shfl_down(m, off));
    m = __shfl(m, 0);
    float s = 0.f;
    for (int c = lane; c < N_; c += 32) {
      float ev = __expf(sS[r][c] - m);
      sS[r][c] = ev;
      s += ev;
    }
    for (int off = 16; off > 0; off >>= 1) s += __shfl_down(s, off);
    s = __shfl(s, 0);
    float inv = 1.f / s;
    for (int c = lane; c < N_; c += 32) sS[r][c] *= inv;
  }
  __syncthreads();
  for (int i = tid; i < N_ * DH_; i += 256) {
    int r = i >> 4, d = i & 15;
    float o = 0.f;
    for (int c = 0; c < N_; ++c) o += sS[r][c] * (float)sV[c][d];
    attn_out[((size_t)b * N_ + r) * E_ + h * 16 + d] = (_Float16)o;
  }
}

// ---------------------------------------------------------------------------
// LayerNorm(residual): y = LN(x + t) * g + b ; writes fp32 + fp16.
// ---------------------------------------------------------------------------
__global__ void ln_kernel(const float* __restrict__ x, const float* __restrict__ t,
                          const float* __restrict__ g, const float* __restrict__ bb,
                          float* __restrict__ h32, _Float16* __restrict__ h16) {
  size_t row = blockIdx.x;
  int e = threadIdx.x;
  __shared__ float red[E_];
  float v = x[row * E_ + e] + t[row * E_ + e];
  red[e] = v;
  __syncthreads();
  for (int s = 64; s > 0; s >>= 1) {
    if (e < s) red[e] += red[e + s];
    __syncthreads();
  }
  float mean = red[0] * (1.0f / E_);
  __syncthreads();
  float d = v - mean;
  red[e] = d * d;
  __syncthreads();
  for (int s = 64; s > 0; s >>= 1) {
    if (e < s) red[e] += red[e + s];
    __syncthreads();
  }
  float var = red[0] * (1.0f / E_);
  float y = d * rsqrtf(var + 1e-5f) * g[e] + bb[e];
  h32[row * E_ + e] = y;
  h16[row * E_ + e] = (_Float16)y;
}

// ---------------------------------------------------------------------------
// Fused FFN: out = relu(h @ ff1.T + b1) @ ff2.T + b2  (no 210MB intermediate).
// h tile DMA'd into LDS by the TDM; each 16x32 ReLU'd FF1 chunk round-trips
// an LDS stage (C-fragment -> A-fragment re-layout) and is consumed
// immediately by FF2 WMMAs accumulating the 16x128 output strip.
// ---------------------------------------------------------------------------
__global__ void ffn_kernel(const _Float16* __restrict__ h16,
                           const _Float16* __restrict__ w1,
                           const float* __restrict__ b1,
                           const _Float16* __restrict__ w2,
                           const float* __restrict__ b2,
                           float* __restrict__ out32) {
  __shared__ _Float16 sH[128 * E_];       // 32 KB
  __shared__ _Float16 sStage[8][16 * 32]; // 8 KB
  int tid = threadIdx.x, wid = tid >> 5, lane = tid & 31;
  int ml = lane & 15, hi = lane >> 4;
  size_t m_block = (size_t)blockIdx.x * 128;
#if USE_TDM
  if (tid < 32) {  // wave 0: one 128x128 fp16 tensor DMA
    tdm_load_2d_f16((unsigned)(uintptr_t)sH, h16 + m_block * E_,
                    E_, 128, E_, BN_, E_);
    __builtin_amdgcn_s_wait_tensorcnt(0);
  }
  __syncthreads();
#else
  for (int i = tid; i < (128 * E_) / 8; i += 256)
    ((v8h*)sH)[i] = ((const v8h*)(h16 + m_block * E_))[i];
  __syncthreads();
#endif

  const _Float16* Abase = sH + (size_t)(wid * 16) * E_;
  _Float16* st = &sStage[wid][0];
  v8f acc[8] = {};

  for (int f0 = 0; f0 < FF_; f0 += 32) {
    // FF1: two 16x16 tiles over K=128
    v8f c1 = {}, c2 = {};
#pragma unroll
    for (int k0 = 0; k0 < E_; k0 += 32) {
      v16h a  = ld_frag(Abase, E_, k0, ml, hi);
      v16h bA = ld_frag(w1 + (size_t)f0 * E_, E_, k0, ml, hi);
      v16h bB = ld_frag(w1 + (size_t)(f0 + 16) * E_, E_, k0, ml, hi);
      c1 = wmma16(a, bA, c1);
      c2 = wmma16(a, bB, c2);
    }
    // bias + relu, re-layout C fragments into LDS stage (row-major 16x32)
#pragma unroll
    for (int r = 0; r < 8; ++r) {
      int mrow = r + hi * 8;
      float x1 = c1[r] + b1[f0 + ml];      x1 = x1 > 0.f ? x1 : 0.f;
      float x2 = c2[r] + b1[f0 + 16 + ml]; x2 = x2 > 0.f ? x2 : 0.f;
      st[mrow * 32 + ml]      = (_Float16)x1;
      st[mrow * 32 + 16 + ml] = (_Float16)x2;
    }
    __syncthreads();
    // FF2: stage is one full K=32 step; accumulate all 8 n-tiles
    v16h af = ld_frag(st, 32, 0, ml, hi);
#pragma unroll
    for (int nt = 0; nt < 8; ++nt) {
      v16h bf = ld_frag(w2 + (size_t)(nt * 16) * FF_ + f0, FF_, 0, ml, hi);
      acc[nt] = wmma16(af, bf, acc[nt]);
    }
    __syncthreads();
  }
#pragma unroll
  for (int nt = 0; nt < 8; ++nt)
#pragma unroll
    for (int r = 0; r < 8; ++r) {
      size_t m = m_block + wid * 16 + r + hi * 8;
      int n = nt * 16 + ml;
      out32[m * E_ + n] = acc[nt][r] + b2[n];
    }
}

// ---------------------------------------------------------------------------
// global_ctx = mean over N
// ---------------------------------------------------------------------------
__global__ void mean_kernel(const float* __restrict__ h32, float* __restrict__ gctx) {
  int b = blockIdx.x, e = threadIdx.x;
  float s = 0.f;
  for (int n = 0; n < N_; ++n) s += h32[((size_t)b * N_ + n) * E_ + e];
  gctx[(size_t)b * E_ + e] = s * (1.0f / N_);
}

// ---------------------------------------------------------------------------
// Greedy decoder: one block per batch element, 199 sequential steps.
// ---------------------------------------------------------------------------
__global__ void decoder_kernel(const float* __restrict__ h32,
                               const float* __restrict__ kall,
                               const float* __restrict__ gctx,
                               const float* __restrict__ dr_all,
                               const float* __restrict__ capW,
                               const float* __restrict__ capb,
                               const float* __restrict__ ctxW,
                               const float* __restrict__ ctxb,
                               const float* __restrict__ qW,
                               float* __restrict__ out) {
  int b = blockIdx.x;
  int e = threadIdx.x;  // 128
  __shared__ float s_qin[3 * E_], s_ctx[E_], s_q[E_], s_logit[N_], s_dr[N_];
  __shared__ int s_vis[N_];
  __shared__ float s_rem, s_lp;
  __shared__ int s_cur, s_done;

  if (e < N_) { s_vis[e] = 0; s_dr[e] = dr_all[(size_t)b * N_ + e]; }
  s_qin[E_ + e] = gctx[(size_t)b * E_ + e];
  if (e == 0) { s_rem = 1.f; s_cur = 0; s_done = 0; s_lp = 0.f; out[(size_t)b * 200] = 0.f; }
  __syncthreads();
  const float invsq = 0.08838834764831845f;  // 1/sqrt(128)

  for (int t = 0; t < 2 * (N_ - 1) + 1; ++t) {
    float rem = s_rem;
    int cur = s_cur;
    s_qin[e] = h32[((size_t)b * N_ + cur) * E_ + e];
    s_qin[2 * E_ + e] = rem * capW[e] + capb[e];
    __syncthreads();
    {
      float a = ctxb[e];
      const float* wr = ctxW + (size_t)e * (3 * E_);
      for (int j = 0; j < 3 * E_; ++j) a += wr[j] * s_qin[j];
      s_ctx[e] = a;
    }
    __syncthreads();
    {
      float a = 0.f;
      const float* wr = qW + (size_t)e * E_;
      for (int j = 0; j < E_; ++j) a += wr[j] * s_ctx[j];
      s_q[e] = a;
    }
    __syncthreads();
    if (e < N_) {
      const float* kr = kall + ((size_t)b * N_ + e) * E_;
      float a = 0.f;
      for (int j = 0; j < E_; ++j) a += kr[j] * s_q[j];
      s_logit[e] = a * invsq;
    }
    __syncthreads();
    if (e == 0) {
      int done = s_done;
      int all_served = 1, has_feas = 0;
      for (int n = 1; n < N_; ++n)
        if (!s_vis[n]) { all_served = 0; if (s_dr[n] <= rem) has_feas = 1; }
      int at_depot = (cur == 0) && !done;
      int force = (all_served && cur != 0 && !done) ||
                  ((cur != 0) && !done && !has_feas) || done;
      float m = -3.4e38f;
      int amax = 0;
      for (int n = 0; n < N_; ++n) {
        int inf_n = s_vis[n] || (s_dr[n] > rem);
        if (n == 0 && at_depot && has_feas) inf_n = 1;
        float l = inf_n ? NEG_ : s_logit[n];
        s_logit[n] = l;
        if (l > m) { m = l; amax = n; }  // first-max, matches jnp.argmax
      }
      float Z = 0.f;
      for (int n = 0; n < N_; ++n) Z += __expf(s_logit[n] - m);
      float lp = __logf(__expf(s_logit[amax] - m) / Z + 1e-12f);
      int sel = amax;
      if (force) { sel = 0; lp = 0.f; }
      if (sel == 0) rem = 1.f; else { rem -= s_dr[sel]; s_vis[sel] = 1; }
      if (all_served && sel == 0) done = 1;
      s_done = done; s_cur = sel; s_rem = rem; s_lp += lp;
      out[(size_t)b * 200 + t + 1] = (float)sel;
    }
    __syncthreads();
  }
  if (e == 0) out[(size_t)B_ * 200 + b] = s_lp;
}

// ---------------------------------------------------------------------------
// Host orchestration
// ---------------------------------------------------------------------------
extern "C" void kernel_launch(void* const* d_in, const int* in_sizes, int n_in,
                              void* d_out, int out_size, void* d_ws, size_t ws_size,
                              hipStream_t stream) {
  (void)in_sizes; (void)n_in; (void)out_size; (void)ws_size;
  // JAX dict-key-sorted flattening of setup_inputs()
  const float* coords   = (const float*)d_in[0];
  const float* demands  = (const float*)d_in[1];
  const float* capacity = (const float*)d_in[2];
  const float* cap_W    = (const float*)d_in[3];
  const float* cap_b    = (const float*)d_in[4];
  const float* ctx_W    = (const float*)d_in[5];
  const float* ctx_b    = (const float*)d_in[6];
  const float* embed_W  = (const float*)d_in[7];
  const float* embed_b  = (const float*)d_in[8];
  const float* k_W      = (const float*)d_in[9];
  const float* q_W      = (const float*)d_in[46];
  struct Layer {
    const float *ff1_W, *ff1_b, *ff2_W, *ff2_b, *in_W, *in_b;
    const float *ln1_b, *ln1_g, *ln2_b, *ln2_g, *out_W, *out_b;
  } L[3];
  for (int l = 0; l < 3; ++l) {
    int base = 10 + 12 * l;
    L[l].ff1_W = (const float*)d_in[base + 0];  L[l].ff1_b = (const float*)d_in[base + 1];
    L[l].ff2_W = (const float*)d_in[base + 2];  L[l].ff2_b = (const float*)d_in[base + 3];
    L[l].in_W  = (const float*)d_in[base + 4];  L[l].in_b  = (const float*)d_in[base + 5];
    L[l].ln1_b = (const float*)d_in[base + 6];  L[l].ln1_g = (const float*)d_in[base + 7];
    L[l].ln2_b = (const float*)d_in[base + 8];  L[l].ln2_g = (const float*)d_in[base + 9];
    L[l].out_W = (const float*)d_in[base + 10]; L[l].out_b = (const float*)d_in[base + 11];
  }

  // workspace carve (256B aligned)
  char* wp = (char*)d_ws;
  auto take = [&](size_t bytes) -> char* {
    char* r = wp; wp += (bytes + 255) & ~(size_t)255; return r;
  };
  float*    h32    = (float*)   take((size_t)BN_ * E_ * 4);
  _Float16* h16    = (_Float16*)take((size_t)BN_ * E_ * 2);
  _Float16* qkv16  = (_Float16*)take((size_t)BN_ * 3 * E_ * 2);
  _Float16* attn16 = (_Float16*)take((size_t)BN_ * E_ * 2);
  float*    tmp32  = (float*)   take((size_t)BN_ * E_ * 4);
  float*    kall32 = (float*)   take((size_t)BN_ * E_ * 4);
  float*    gctx   = (float*)   take((size_t)B_ * E_ * 4);
  float*    dr     = (float*)   take((size_t)BN_ * 4);
  _Float16* inW16[3];  _Float16* outW16[3];
  _Float16* ff1W16[3]; _Float16* ff2W16[3];
  for (int l = 0; l < 3; ++l) {
    inW16[l]  = (_Float16*)take((size_t)3 * E_ * E_ * 2);
    outW16[l] = (_Float16*)take((size_t)E_ * E_ * 2);
    ff1W16[l] = (_Float16*)take((size_t)FF_ * E_ * 2);
    ff2W16[l] = (_Float16*)take((size_t)E_ * FF_ * 2);
  }
  _Float16* kW16 = (_Float16*)take((size_t)E_ * E_ * 2);

  auto cvt = [&](const float* s, _Float16* d, int n) {
    cvt_f16_kernel<<<(n + 255) / 256, 256, 0, stream>>>(s, d, n);
  };
  for (int l = 0; l < 3; ++l) {
    cvt(L[l].in_W,  inW16[l],  3 * E_ * E_);
    cvt(L[l].out_W, outW16[l], E_ * E_);
    cvt(L[l].ff1_W, ff1W16[l], FF_ * E_);
    cvt(L[l].ff2_W, ff2W16[l], E_ * FF_);
  }
  cvt(k_W, kW16, E_ * E_);

  embed_kernel<<<(BN_ * E_ + 255) / 256, 256, 0, stream>>>(
      coords, demands, capacity, embed_W, embed_b, h32, h16, dr);

  const int MB = BN_ / 128;  // 400 row-blocks
  for (int l = 0; l < 3; ++l) {
    // QKV projection (fp16 out)
    gemm_kernel<<<dim3(MB, (3 * E_) / 64), 256, 0, stream>>>(
        h16, inW16[l], L[l].in_b, (float*)nullptr, qkv16, 3 * E_, E_, 0);
    // attention
    attn_kernel<<<B_ * H_, 256, 0, stream>>>(qkv16, attn16);
    // out projection (fp32 out)
    gemm_kernel<<<dim3(MB, E_ / 64), 256, 0, stream>>>(
        attn16, outW16[l], L[l].out_b, tmp32, (_Float16*)nullptr, E_, E_, 0);
    // x = LN1(x + proj)
    ln_kernel<<<BN_, E_, 0, stream>>>(h32, tmp32, L[l].ln1_g, L[l].ln1_b, h32, h16);
    // fused FFN
    ffn_kernel<<<MB, 256, 0, stream>>>(h16, ff1W16[l], L[l].ff1_b,
                                       ff2W16[l], L[l].ff2_b, tmp32);
    // x = LN2(x + ffn)
    ln_kernel<<<BN_, E_, 0, stream>>>(h32, tmp32, L[l].ln2_g, L[l].ln2_b, h32, h16);
  }

  // k_all = h @ k_W.T (no bias), global context mean
  gemm_kernel<<<dim3(MB, E_ / 64), 256, 0, stream>>>(
      h16, kW16, (const float*)nullptr, kall32, (_Float16*)nullptr, E_, E_, 0);
  mean_kernel<<<B_, E_, 0, stream>>>(h32, gctx);

  decoder_kernel<<<B_, E_, 0, stream>>>(h32, kall32, gctx, dr, cap_W, cap_b,
                                        ctx_W, ctx_b, q_W, (float*)d_out);
}
